// Quantized_37211596652827
// MI455X (gfx1250) — compile-verified, matrix operand order checked by reading, not run on hardware
//
#include <hip/hip_runtime.h>
#include <hip/hip_bf16.h>

typedef __attribute__((ext_vector_type(16))) __bf16 v16bf;
typedef __attribute__((ext_vector_type(8)))  __bf16 v8bf;
typedef __attribute__((ext_vector_type(4)))  __bf16 v4bf;
typedef __attribute__((ext_vector_type(8)))  float  v8f;

#define BM 256
#define BN 256
#define BK 32
#define LDT 40   // padded LDS row stride (elements): 80B = 20 banks -> conflict-free b128 frag loads
#define N_GROUPS 16
#define NTHREADS 512

__global__ __launch_bounds__(NTHREADS) void qgemm_wmma_bf16(
    const float* __restrict__ x, const int* __restrict__ qw,
    const float* __restrict__ qrange, const float* __restrict__ qmin,
    float* __restrict__ out, int M, int N, int K)
{
    // double-buffered tiles: 2 x 256x40 bf16 each = 40KB per array, 80KB total
    __shared__ __align__(16) __bf16 As[2][BM * LDT];   // x tile (m x k), bf16
    __shared__ __align__(16) __bf16 Bs[2][BN * LDT];   // w tile (n x k), dequantized bf16

    const int tid  = threadIdx.x;
    const int lane = tid & 31;
    const int wave = tid >> 5;        // 0..15
    const int wm   = wave & 3;        // wave M position (0..3), 64 rows each
    const int wn   = wave >> 2;       // wave N position (0..3), 64 cols each
    const int m0   = blockIdx.y * BM;
    const int n0   = blockIdx.x * BN;

    const int lrow  = lane & 15;          // row within 16x16 tile
    const int khalf = (lane >> 4) << 3;   // 0 or 8: K-half select per ISA A/B layout

    v8f acc[4][4];
#pragma unroll
    for (int i = 0; i < 4; ++i)
#pragma unroll
        for (int j = 0; j < 4; ++j)
            acc[i][j] = (v8f){0.f,0.f,0.f,0.f,0.f,0.f,0.f,0.f};

    const int gsz = K / N_GROUPS;         // 256
    const float inv255 = 1.0f / 255.0f;
    const int KT = K / BK;                // 128 k-tiles

    // ---- per-thread loader geometry: 512 threads x 4 slots cover 256 rows x 8 float4
    const float* xptr[4];
    const int*   qptr[4];
    int nrow16[4];        // n * N_GROUPS for scale lookups
    int ldsoff[4];        // LDS destination offset (same indexing for As and Bs)
#pragma unroll
    for (int i = 0; i < 4; ++i) {
        const int idx = tid + i * NTHREADS;
        const int r   = idx >> 3;           // 0..255
        const int c4  = (idx & 7) << 2;     // 0,4,...,28
        xptr[i]   = x  + (size_t)(m0 + r) * K + c4;
        qptr[i]   = qw + (size_t)(n0 + r) * K + c4;
        nrow16[i] = (n0 + r) * N_GROUPS;
        ldsoff[i] = r * LDT + c4;
    }

    // ---- prologue: fill buffer 0 with k-tile 0
    {
        float4 xs[4]; int4 qs[4]; float ss[4], bb[4];
#pragma unroll
        for (int i = 0; i < 4; ++i) {
            xs[i] = *(const float4*)(xptr[i]);
            qs[i] = *(const int4*)(qptr[i]);
            ss[i] = qrange[nrow16[i]] * inv255;   // g = 0
            bb[i] = qmin[nrow16[i]];
        }
#pragma unroll
        for (int i = 0; i < 4; ++i) {
            v4bf av;
            av[0] = (__bf16)xs[i].x; av[1] = (__bf16)xs[i].y;
            av[2] = (__bf16)xs[i].z; av[3] = (__bf16)xs[i].w;
            *(v4bf*)(&As[0][ldsoff[i]]) = av;
            v4bf bv;
            bv[0] = (__bf16)fmaf((float)qs[i].x, ss[i], bb[i]);
            bv[1] = (__bf16)fmaf((float)qs[i].y, ss[i], bb[i]);
            bv[2] = (__bf16)fmaf((float)qs[i].z, ss[i], bb[i]);
            bv[3] = (__bf16)fmaf((float)qs[i].w, ss[i], bb[i]);
            *(v4bf*)(&Bs[0][ldsoff[i]]) = bv;
        }
    }
    __syncthreads();

    for (int kt = 0; kt < KT; ++kt) {
        const int cur  = kt & 1;
        const int nxt  = cur ^ 1;
        const bool have_next = (kt + 1 < KT);

        // ---- issue ALL global loads for tile kt+1 (independent dests -> all in flight)
        float4 xs[4]; int4 qs[4]; float ss[4], bb[4];
        if (have_next) {
            const int koff = (kt + 1) * BK;
            const int g    = koff / gsz;
#pragma unroll
            for (int i = 0; i < 4; ++i) {
                xs[i] = *(const float4*)(xptr[i] + koff);
                qs[i] = *(const int4*)(qptr[i] + koff);
                ss[i] = qrange[nrow16[i] + g] * inv255;
                bb[i] = qmin[nrow16[i] + g];
            }
            if (kt + 2 < KT) {   // L2 prefetch two tiles ahead
                const int pk = (kt + 2) * BK;
                __builtin_prefetch(xptr[tid & 3] + pk, 0, 1);
                __builtin_prefetch(qptr[tid & 3] + pk, 0, 1);
            }
        }

        // ---- compute on buffer `cur`: 4 A-frags held, B-frags streamed (16 WMMAs)
        v16bf af[4];
#pragma unroll
        for (int i = 0; i < 4; ++i) {
            const int r = wm * 64 + i * 16 + lrow;
            const v8bf lo = *(const v8bf*)(&As[cur][r * LDT + khalf]);
            const v8bf hi = *(const v8bf*)(&As[cur][r * LDT + khalf + 16]);
#pragma unroll
            for (int t = 0; t < 8; ++t) { af[i][t] = lo[t]; af[i][t + 8] = hi[t]; }
        }
#pragma unroll
        for (int j = 0; j < 4; ++j) {
            const int r = wn * 64 + j * 16 + lrow;
            const v8bf lo = *(const v8bf*)(&Bs[cur][r * LDT + khalf]);
            const v8bf hi = *(const v8bf*)(&Bs[cur][r * LDT + khalf + 16]);
            v16bf bfrag;
#pragma unroll
            for (int t = 0; t < 8; ++t) { bfrag[t] = lo[t]; bfrag[t + 8] = hi[t]; }
#pragma unroll
            for (int i = 0; i < 4; ++i)
                acc[i][j] = __builtin_amdgcn_wmma_f32_16x16x32_bf16(
                    false, af[i], false, bfrag, (short)0, acc[i][j], false, false);
        }

        // ---- convert staged tile kt+1 -> buffer `nxt`
        if (have_next) {
#pragma unroll
            for (int i = 0; i < 4; ++i) {
                v4bf av;
                av[0] = (__bf16)xs[i].x; av[1] = (__bf16)xs[i].y;
                av[2] = (__bf16)xs[i].z; av[3] = (__bf16)xs[i].w;
                *(v4bf*)(&As[nxt][ldsoff[i]]) = av;
                v4bf bv;
                bv[0] = (__bf16)fmaf((float)qs[i].x, ss[i], bb[i]);
                bv[1] = (__bf16)fmaf((float)qs[i].y, ss[i], bb[i]);
                bv[2] = (__bf16)fmaf((float)qs[i].z, ss[i], bb[i]);
                bv[3] = (__bf16)fmaf((float)qs[i].w, ss[i], bb[i]);
                *(v4bf*)(&Bs[nxt][ldsoff[i]]) = bv;
            }
        }
        __syncthreads();
    }

    // ---- epilogue: C/D layout -> lanes 0-15: M=v, N=lane; lanes 16-31: M=v+8, N=lane-16
    const int mbase = (lane >> 4) << 3;
#pragma unroll
    for (int i = 0; i < 4; ++i) {
#pragma unroll
        for (int j = 0; j < 4; ++j) {
            const int nn = n0 + wn * 64 + j * 16 + lrow;
#pragma unroll
            for (int v = 0; v < 8; ++v) {
                const int mm = m0 + wm * 64 + i * 16 + mbase + v;
                out[(size_t)mm * N + nn] = acc[i][j][v];
            }
        }
    }
}

extern "C" void kernel_launch(void* const* d_in, const int* in_sizes, int n_in,
                              void* d_out, int out_size, void* d_ws, size_t ws_size,
                              hipStream_t stream) {
    const float* x      = (const float*)d_in[0];
    const int*   qw     = (const int*)d_in[1];
    const float* qrange = (const float*)d_in[2];
    const float* qmin   = (const float*)d_in[3];
    float* out = (float*)d_out;

    const long long N = (long long)in_sizes[2] / N_GROUPS;         // 4096
    const long long K = (long long)in_sizes[1] / N;                // 4096
    const long long M = (long long)in_sizes[0] / K;                // 8192

    dim3 grid((unsigned)(N / BN), (unsigned)(M / BM));
    qgemm_wmma_bf16<<<grid, NTHREADS, 0, stream>>>(x, qw, qrange, qmin, out,
                                                   (int)M, (int)N, (int)K);
}
